// SAGRN_40896678593044
// MI455X (gfx1250) — compile-verified, matrix operand order checked by reading, not run on hardware
//
#include <hip/hip_runtime.h>
#include <hip/hip_bf16.h>

// ---------------------------------------------------------------------------
// SAGRN: superpixel graph attention.  Sizes fixed by the reference.
// ---------------------------------------------------------------------------
#define CC    128
#define C4    32
#define HH    512
#define WW    512
#define HWPX  (HH * WW)          // 262144
#define SPN   256
#define NTILES (HWPX / 16)       // 16384

typedef __attribute__((ext_vector_type(16))) __bf16 v16bf;
typedef __attribute__((ext_vector_type(8)))  __bf16 v8bf;
typedef __attribute__((ext_vector_type(8)))  float  v8f;

__device__ __forceinline__ v8f wmma_bf16(v16bf a, v16bf b, v8f c) {
    return __builtin_amdgcn_wmma_f32_16x16x32_bf16(
        /*neg_a=*/false, a, /*neg_b=*/false, b,
        /*c_mod=*/(short)0, c, /*reuse_a=*/false, /*reuse_b=*/false);
}

// ---------------- workspace layout (float offsets) -------------------------
#define OFF_SUMS   0          // 256*128
#define OFF_CNTS   32768      // 256
#define OFF_QG     33024      // 256*32
#define OFF_KG     41216      // 256*32
#define OFF_VG     49408      // 256*128
#define OFF_SUP    82176      // 256*128
#define OFF_ADJ    114944     // 256*256
#define OFF_XPOS   180480     // 256*128
#define OFF_Q2     213248     // 256*32
#define OFF_V2     221440     // 256*128
#define OFF_PACK_BYTES (254208u * 4u)   // bf16 fragment region (32B aligned)
#define PACKA_FRAGS 8     // a_wk^T : 4 kc x 2 nt
#define PACKQ_FRAGS 16    // q2^T   : 1 kc x 16 nt
#define PACKV_FRAGS 64    // v2     : 8 kc x 8 nt

// ---------------------------------------------------------------------------
__global__ void sagrn_zero(float* buf) {
    int i = blockIdx.x * 256 + threadIdx.x;
    if (i < SPN * CC + SPN) buf[i] = 0.0f;
}

// scatter-mean accumulation with run merging (superpixels are spatially
// coherent, so 8 consecutive pixels usually share one label).
__global__ void sagrn_scatter(const float* __restrict__ x, const int* __restrict__ sp,
                              float* __restrict__ sums, float* __restrict__ cnts) {
    int t = blockIdx.x * 256 + threadIdx.x;
    int p0 = t * 8;
    if (p0 >= HWPX) return;
    int lab0 = sp[p0];
    bool uni = true;
#pragma unroll
    for (int j = 1; j < 8; ++j) uni = uni && (sp[p0 + j] == lab0);
    if (uni) {
        atomicAdd(&cnts[lab0], 8.0f);
        float* row = sums + (size_t)lab0 * CC;
        for (int c = 0; c < CC; ++c) {
            const float* px = x + (size_t)c * HWPX + p0;
            float s = 0.0f;
#pragma unroll
            for (int j = 0; j < 8; ++j) s += px[j];
            atomicAdd(&row[c], s);
        }
    } else {
#pragma unroll 1
        for (int j = 0; j < 8; ++j) {
            int lab = sp[p0 + j];
            atomicAdd(&cnts[lab], 1.0f);
            float* row = sums + (size_t)lab * CC;
            for (int c = 0; c < CC; ++c)
                atomicAdd(&row[c], x[(size_t)c * HWPX + p0 + j]);
        }
    }
}

// sp_mean -> q,k,v   (one block of 256 threads, thread = superpixel row)
__global__ void sagrn_meanproj(const float* __restrict__ sums, const float* __restrict__ cnts,
                               const float* __restrict__ wq, const float* __restrict__ bq,
                               const float* __restrict__ wk, const float* __restrict__ bk,
                               const float* __restrict__ wv, const float* __restrict__ bv,
                               float* __restrict__ qg, float* __restrict__ kg,
                               float* __restrict__ vg) {
    int i = threadIdx.x;
    float inv = 1.0f / fmaxf(cnts[i], 1.0f);
    float mr[CC];
    for (int c = 0; c < CC; ++c) mr[c] = sums[i * CC + c] * inv;
    for (int j = 0; j < C4; ++j) {
        float aq = bq[j], ak = bk[j];
        for (int c = 0; c < CC; ++c) {
            aq += mr[c] * wq[j * CC + c];
            ak += mr[c] * wk[j * CC + c];
        }
        qg[i * C4 + j] = aq;
        kg[i * C4 + j] = ak;
    }
    for (int j = 0; j < CC; ++j) {
        float av = bv[j];
        for (int c = 0; c < CC; ++c) av += mr[c] * wv[j * CC + c];
        vg[i * CC + j] = av;
    }
}

// support = v @ gcn_weight
__global__ void sagrn_support(const float* __restrict__ vg, const float* __restrict__ gw,
                              float* __restrict__ sup) {
    int i = threadIdx.x;
    for (int cc = 0; cc < 4; ++cc) {
        float acc[32];
#pragma unroll
        for (int t = 0; t < 32; ++t) acc[t] = 0.0f;
        for (int c = 0; c < CC; ++c) {
            float vv = vg[i * CC + c];
            const float* wrow = gw + c * CC + cc * 32;
#pragma unroll
            for (int t = 0; t < 32; ++t) acc[t] += vv * wrow[t];
        }
#pragma unroll
        for (int t = 0; t < 32; ++t) sup[i * CC + cc * 32 + t] = acc[t];
    }
}

// adj = softmax(k @ q.T); xpos = adj @ support + gcn_bias
__global__ void sagrn_adjxpos(const float* __restrict__ kg, const float* __restrict__ qg,
                              const float* __restrict__ sup, const float* __restrict__ gb,
                              float* __restrict__ adj, float* __restrict__ xpos) {
    int i = threadIdx.x;
    float kr[C4];
    for (int c = 0; c < C4; ++c) kr[c] = kg[i * C4 + c];
    float mx = -3.0e38f;
    for (int j = 0; j < SPN; ++j) {
        float d = 0.0f;
        for (int c = 0; c < C4; ++c) d += kr[c] * qg[j * C4 + c];
        adj[i * SPN + j] = d;
        mx = fmaxf(mx, d);
    }
    float s = 0.0f;
    for (int j = 0; j < SPN; ++j) {
        float e = __expf(adj[i * SPN + j] - mx);
        adj[i * SPN + j] = e;
        s += e;
    }
    float inv = 1.0f / s;
    for (int cc = 0; cc < 4; ++cc) {
        float acc[32];
#pragma unroll
        for (int t = 0; t < 32; ++t) acc[t] = 0.0f;
        for (int j = 0; j < SPN; ++j) {
            float w = adj[i * SPN + j];
            const float* srow = sup + j * CC + cc * 32;
#pragma unroll
            for (int t = 0; t < 32; ++t) acc[t] += w * srow[t];
        }
#pragma unroll
        for (int t = 0; t < 32; ++t)
            xpos[i * CC + cc * 32 + t] = acc[t] * inv + gb[cc * 32 + t];
    }
}

// q2 = xpos @ a_wq.T + a_bq ; v2 = xpos @ a_wv.T + a_bv
__global__ void sagrn_proj2(const float* __restrict__ xpos,
                            const float* __restrict__ awq, const float* __restrict__ abq,
                            const float* __restrict__ awv, const float* __restrict__ abv,
                            float* __restrict__ q2, float* __restrict__ v2) {
    int i = threadIdx.x;
    float xr[CC];
    for (int c = 0; c < CC; ++c) xr[c] = xpos[i * CC + c];
    for (int j = 0; j < C4; ++j) {
        float a = abq[j];
        for (int c = 0; c < CC; ++c) a += xr[c] * awq[j * CC + c];
        q2[i * C4 + j] = a;
    }
    for (int j = 0; j < CC; ++j) {
        float a = abv[j];
        for (int c = 0; c < CC; ++c) a += xr[c] * awv[j * CC + c];
        v2[i * CC + j] = a;
    }
}

// Pack the shared B operands into per-lane bf16 WMMA fragments.
// 16-bit B layout (32x16, K-major pairs):
//   lane<16 : col N = lane, elems 0..7 = K(kg*8+0..7), 8..15 = K(kg*8+16..23)
__global__ void sagrn_pack(const float* __restrict__ a_wk, const float* __restrict__ q2,
                           const float* __restrict__ v2, __bf16* __restrict__ pA,
                           __bf16* __restrict__ pQ, __bf16* __restrict__ pV) {
    int frag = blockIdx.x;
    int lane = threadIdx.x;
    int m = lane & 15, kg = lane >> 4;
    if (frag < PACKA_FRAGS) {                 // B = a_wk^T  (K=128, N=32)
        int kc = frag >> 1, nt = frag & 1;
        __bf16* dst = pA + ((size_t)frag * 32 + lane) * 16;
        const float* p = a_wk + (nt * 16 + m) * CC + kc * 32 + kg * 8;
#pragma unroll
        for (int i = 0; i < 8; ++i) { dst[i] = (__bf16)p[i]; dst[8 + i] = (__bf16)p[16 + i]; }
    } else if (frag < PACKA_FRAGS + PACKQ_FRAGS) {  // B = q2^T (K=32, N=256)
        int nt = frag - PACKA_FRAGS;
        __bf16* dst = pQ + ((size_t)nt * 32 + lane) * 16;
        const float* p = q2 + (nt * 16 + m) * C4 + kg * 8;
#pragma unroll
        for (int i = 0; i < 8; ++i) { dst[i] = (__bf16)p[i]; dst[8 + i] = (__bf16)p[16 + i]; }
    } else {                                   // B = v2 (K=256, N=128), row-major K x N
        int f = frag - PACKA_FRAGS - PACKQ_FRAGS;
        int kc = f >> 3, nt = f & 7;
        __bf16* dst = pV + ((size_t)f * 32 + lane) * 16;
        const float* p = v2 + (size_t)(kc * 32 + kg * 8) * CC + nt * 16 + m;
#pragma unroll
        for (int i = 0; i < 8; ++i) {
            dst[i]     = (__bf16)p[(size_t)i * CC];
            dst[8 + i] = (__bf16)p[(size_t)(16 + i) * CC];
        }
    }
}

// ---------------------------------------------------------------------------
// Fused cross attention, one wave per 16-pixel tile.
//  - x tile (128ch x 16pix f32) staged via global_load_async_to_lds_b128
//  - k2 = feat@a_wk.T + bk         (8 wmma)
//  - logits = k2@q2.T, two 128-col blocks, online-softmax rescale (16 wmma)
//  - atten stored bf16 in LDS; out = atten@v2 (64 wmma), per-row f/inv scaling
//  - LDS transpose -> coalesced b128 stores into out[C][HW]
// Per-wave LDS (float slots): [0..2047] XT / ATT(bf16) / OUT (sequential reuse)
//                             [2048..2559] k2 tile.  stride 2688 -> 43KB/block
// ---------------------------------------------------------------------------
__global__ __launch_bounds__(128) void sagrn_attn(
        const float* __restrict__ x, const __bf16* __restrict__ packA,
        const __bf16* __restrict__ packQ, const __bf16* __restrict__ packV,
        const float* __restrict__ a_bk, float* __restrict__ out) {
    __shared__ float lds[4 * 2688];
    const int lane  = threadIdx.x & 31;
    const int wslot = threadIdx.x >> 5;
    float* sm   = lds + wslot * 2688;
    float* smK2 = sm + 2048;
    __bf16* att = (__bf16*)sm;
    const int m  = lane & 15;
    const int hi = lane >> 4;
    const int nwaves = gridDim.x * 4;
    const int wid = blockIdx.x * 4 + wslot;

    const v16bf* fA = (const v16bf*)packA;
    const v16bf* fQ = (const v16bf*)packQ;
    const v16bf* fV = (const v16bf*)packV;
    const float bk0 = a_bk[m];
    const float bk1 = a_bk[16 + m];
    const v8f z = {0.f, 0.f, 0.f, 0.f, 0.f, 0.f, 0.f, 0.f};

    for (int tile = wid; tile < NTILES; tile += nwaves) {
        const int pix0 = tile * 16;
        {   // prefetch next tile (global_prefetch_b8)
            int np = pix0 + nwaves * 16;
            if (np < HWPX) __builtin_prefetch(x + (size_t)m * HWPX + np, 0, 1);
        }
        // ---- stage x tile: LDS[ch][0..15] = x[ch][pix0..pix0+15], async b128
#pragma unroll
        for (int i = 0; i < 16; ++i) {
            int ch = i * 8 + (lane >> 2);
            int pg = (lane & 3) * 4;
            const float* gp = x + (size_t)ch * HWPX + pix0 + pg;
            unsigned lo = (unsigned)(uintptr_t)(const void*)(sm + ch * 16 + pg);
            asm volatile("global_load_async_to_lds_b128 %0, %1, off"
                         :: "v"(lo), "v"(gp) : "memory");
        }
        asm volatile("s_wait_asynccnt 0x0" ::: "memory");

        // ---- k2 = feat @ a_wk.T  (16x32), A-frags built from staged tile
        v8f k20 = z, k21 = z;
#pragma unroll
        for (int kc = 0; kc < 4; ++kc) {
            v16bf a;
            const float* p = sm + (kc * 32 + hi * 8) * 16 + m;
#pragma unroll
            for (int i = 0; i < 8; ++i) {
                a[i]     = (__bf16)p[i * 16];
                a[8 + i] = (__bf16)p[(16 + i) * 16];
            }
            k20 = wmma_bf16(a, fA[(kc * 2 + 0) * 32 + lane], k20);
            k21 = wmma_bf16(a, fA[(kc * 2 + 1) * 32 + lane], k21);
        }
        // bias + spill as 16x32 (wave-synchronous transpose)
#pragma unroll
        for (int r = 0; r < 8; ++r) {
            smK2[(r + 8 * hi) * 32 + m]      = k20[r] + bk0;
            smK2[(r + 8 * hi) * 32 + 16 + m] = k21[r] + bk1;
        }
        v16bf ak;
        {
            const float* p = smK2 + m * 32 + hi * 8;
#pragma unroll
            for (int i = 0; i < 8; ++i) { ak[i] = (__bf16)p[i]; ak[8 + i] = (__bf16)p[16 + i]; }
        }

        // ---- logits block A (superpixels 0..127)
        v8f lg[8];
#pragma unroll
        for (int nt = 0; nt < 8; ++nt) lg[nt] = wmma_bf16(ak, fQ[nt * 32 + lane], z);
        float mA[8];
#pragma unroll
        for (int r = 0; r < 8; ++r) {
            float v = lg[0][r];
#pragma unroll
            for (int nt = 1; nt < 8; ++nt) v = fmaxf(v, lg[nt][r]);
            mA[r] = v;
        }
#pragma unroll
        for (int off = 1; off < 16; off <<= 1)
#pragma unroll
            for (int r = 0; r < 8; ++r) mA[r] = fmaxf(mA[r], __shfl_xor(mA[r], off, 16));
        float sA[8];
#pragma unroll
        for (int r = 0; r < 8; ++r) sA[r] = 0.0f;
#pragma unroll
        for (int nt = 0; nt < 8; ++nt)
#pragma unroll
            for (int r = 0; r < 8; ++r) {
                float e = __expf(lg[nt][r] - mA[r]);
                sA[r] += e;
                att[(r + 8 * hi) * 256 + nt * 16 + m] = (__bf16)e;
            }
#pragma unroll
        for (int off = 1; off < 16; off <<= 1)
#pragma unroll
            for (int r = 0; r < 8; ++r) sA[r] += __shfl_xor(sA[r], off, 16);

        // ---- logits block B (superpixels 128..255), online rescale of A
#pragma unroll
        for (int nt = 0; nt < 8; ++nt) lg[nt] = wmma_bf16(ak, fQ[(8 + nt) * 32 + lane], z);
        float mB[8];
#pragma unroll
        for (int r = 0; r < 8; ++r) {
            float v = lg[0][r];
#pragma unroll
            for (int nt = 1; nt < 8; ++nt) v = fmaxf(v, lg[nt][r]);
            mB[r] = v;
        }
#pragma unroll
        for (int off = 1; off < 16; off <<= 1)
#pragma unroll
            for (int r = 0; r < 8; ++r) mB[r] = fmaxf(mB[r], __shfl_xor(mB[r], off, 16));
        float M2[8], fcor[8], sB[8];
#pragma unroll
        for (int r = 0; r < 8; ++r) {
            M2[r] = fmaxf(mA[r], mB[r]);
            fcor[r] = __expf(mA[r] - M2[r]);   // <= 1
            sB[r] = 0.0f;
        }
#pragma unroll
        for (int nt = 0; nt < 8; ++nt)
#pragma unroll
            for (int r = 0; r < 8; ++r) {
                float e = __expf(lg[nt][r] - M2[r]);
                sB[r] += e;
                att[(r + 8 * hi) * 256 + (8 + nt) * 16 + m] = (__bf16)e;
            }
#pragma unroll
        for (int off = 1; off < 16; off <<= 1)
#pragma unroll
            for (int r = 0; r < 8; ++r) sB[r] += __shfl_xor(sB[r], off, 16);
        float inv[8];
#pragma unroll
        for (int r = 0; r < 8; ++r)
            inv[r] = __builtin_amdgcn_rcpf(sA[r] * fcor[r] + sB[r]);

        // ---- out = atten @ v2, block-A part then rescale, then block-B part
        v8f oc[8];
#pragma unroll
        for (int nt = 0; nt < 8; ++nt) oc[nt] = z;
#pragma unroll
        for (int kc = 0; kc < 4; ++kc) {
            const __bf16* ap = att + m * 256 + kc * 32 + hi * 8;
            v8bf loh = *(const v8bf*)ap;
            v8bf hih = *(const v8bf*)(ap + 16);
            v16bf a = __builtin_shufflevector(loh, hih, 0, 1, 2, 3, 4, 5, 6, 7,
                                              8, 9, 10, 11, 12, 13, 14, 15);
#pragma unroll
            for (int nt = 0; nt < 8; ++nt)
                oc[nt] = wmma_bf16(a, fV[(kc * 8 + nt) * 32 + lane], oc[nt]);
        }
#pragma unroll
        for (int nt = 0; nt < 8; ++nt)
#pragma unroll
            for (int r = 0; r < 8; ++r) oc[nt][r] *= fcor[r];
#pragma unroll
        for (int kc = 4; kc < 8; ++kc) {
            const __bf16* ap = att + m * 256 + kc * 32 + hi * 8;
            v8bf loh = *(const v8bf*)ap;
            v8bf hih = *(const v8bf*)(ap + 16);
            v16bf a = __builtin_shufflevector(loh, hih, 0, 1, 2, 3, 4, 5, 6, 7,
                                              8, 9, 10, 11, 12, 13, 14, 15);
#pragma unroll
            for (int nt = 0; nt < 8; ++nt)
                oc[nt] = wmma_bf16(a, fV[(kc * 8 + nt) * 32 + lane], oc[nt]);
        }
        // stage [16 pix][128 ch] f32 (region reuse after atten consumed)
#pragma unroll
        for (int nt = 0; nt < 8; ++nt)
#pragma unroll
            for (int r = 0; r < 8; ++r)
                sm[(r + 8 * hi) * 128 + nt * 16 + m] = oc[nt][r] * inv[r];
        // coalesced b128 stores, out is [C][HW]
#pragma unroll
        for (int it = 0; it < 16; ++it) {
            int ch = it * 8 + (lane >> 2);
            int pg = (lane & 3) * 4;
            float4 vv;
            vv.x = sm[(pg + 0) * 128 + ch];
            vv.y = sm[(pg + 1) * 128 + ch];
            vv.z = sm[(pg + 2) * 128 + ch];
            vv.w = sm[(pg + 3) * 128 + ch];
            *(float4*)(out + (size_t)ch * HWPX + pix0 + pg) = vv;
        }
    }
}

// ---------------------------------------------------------------------------
extern "C" void kernel_launch(void* const* d_in, const int* in_sizes, int n_in,
                              void* d_out, int out_size, void* d_ws, size_t ws_size,
                              hipStream_t stream) {
    const float* x     = (const float*)d_in[0];
    const int*   sp    = (const int*)  d_in[1];
    const float* gcn_w = (const float*)d_in[2];
    const float* gcn_b = (const float*)d_in[3];
    const float* g_wq  = (const float*)d_in[4];
    const float* g_bq  = (const float*)d_in[5];
    const float* g_wk  = (const float*)d_in[6];
    const float* g_bk  = (const float*)d_in[7];
    const float* g_wv  = (const float*)d_in[8];
    const float* g_bv  = (const float*)d_in[9];
    const float* a_wq  = (const float*)d_in[10];
    const float* a_bq  = (const float*)d_in[11];
    const float* a_wk  = (const float*)d_in[12];
    const float* a_bk  = (const float*)d_in[13];
    const float* a_wv  = (const float*)d_in[14];
    const float* a_bv  = (const float*)d_in[15];

    float* ws   = (float*)d_ws;
    float* sums = ws + OFF_SUMS;
    float* cnts = ws + OFF_CNTS;
    float* qg   = ws + OFF_QG;
    float* kg   = ws + OFF_KG;
    float* vg   = ws + OFF_VG;
    float* sup  = ws + OFF_SUP;
    float* adj  = ws + OFF_ADJ;
    float* xpos = ws + OFF_XPOS;
    float* q2   = ws + OFF_Q2;
    float* v2   = ws + OFF_V2;
    __bf16* pA = (__bf16*)((char*)d_ws + OFF_PACK_BYTES);
    __bf16* pQ = pA + (size_t)PACKA_FRAGS * 32 * 16;
    __bf16* pV = pQ + (size_t)PACKQ_FRAGS * 32 * 16;

    float* outp = (float*)d_out;
    const size_t CHW = (size_t)CC * HWPX;

    sagrn_zero<<<(SPN * CC + SPN + 255) / 256, 256, 0, stream>>>(sums);
    sagrn_scatter<<<HWPX / 8 / 256, 256, 0, stream>>>(x, sp, sums, cnts);
    sagrn_meanproj<<<1, 256, 0, stream>>>(sums, cnts, g_wq, g_bq, g_wk, g_bk,
                                          g_wv, g_bv, qg, kg, vg);
    sagrn_support<<<1, 256, 0, stream>>>(vg, gcn_w, sup);
    sagrn_adjxpos<<<1, 256, 0, stream>>>(kg, qg, sup, gcn_b, adj, xpos);
    sagrn_proj2<<<1, 256, 0, stream>>>(xpos, a_wq, a_bq, a_wv, a_bv, q2, v2);
    sagrn_pack<<<PACKA_FRAGS + PACKQ_FRAGS + PACKV_FRAGS, 32, 0, stream>>>(
        a_wk, q2, v2, pA, pQ, pV);
    sagrn_attn<<<512, 128, 0, stream>>>(x, pA, pQ, pV, a_bk, outp);

    // reference returns (out, x): copy x into second half of d_out
    hipMemcpyAsync(outp + CHW, x, CHW * sizeof(float),
                   hipMemcpyDeviceToDevice, stream);
}
